// GInvar_EncOpt_56401510531403
// MI455X (gfx1250) — compile-verified, matrix operand order checked by reading, not run on hardware
//
#include <hip/hip_runtime.h>
#include <hip/hip_bf16.h>
#include <math.h>

typedef __attribute__((ext_vector_type(16))) _Float16 v16h;
typedef __attribute__((ext_vector_type(8)))  _Float16 v8h;
typedef __attribute__((ext_vector_type(8)))  float    v8f;

#define NN 8
#define HH 4
#define DOUT 2
#define DIN 256
#define BATCH_PER_BLOCK 16   // 8 waves * 2 batches per wave
#define ROWS_PER_BLOCK  128  // 16 batches * 8 nodes

// ---- const image layout in d_ws -------------------------------------------
// float region (indices into float*):
#define OFF_AN   0      // 64
#define OFF_AC   64     // 64
#define OFF_AA   128    // 64
#define OFF_CB   192    // 4
#define OFF_V1B  196    // 4
#define OFF_V2B  200    // 4
#define OFF_V3B  204    // 4
#define OFF_GW   208    // 96
#define OFF_GB   304    // 3
#define OFF_MUW  308    // 8
#define OFF_MUB  316    // 2
#define OFF_LSW  318    // 8
#define OFF_LSB  326    // 2
#define CONST_FLOATS 328
// f16 region starts at float index 512 (byte offset 2048, 16B aligned):
#define F16_BYTE_OFF  2048
#define WPACK_HALVES  (2 * 16 * DIN)   // WcatT[4096] then WnoiseT[4096]

// noise value scale folded between hash output and weights:
//   noise elem = (float)(int16) * 2^-10   (range ~ +-32, normal f16)
//   Wnoise    = (0.1 * 2^10 / 32768) * v1_W = (0.1/32) * v1_W
#define NOISE_ELEM_SCALE 0x1p-10f
#define NOISE_W_SCALE    (0.1f / 32.0f)

__device__ __forceinline__ unsigned hash32(unsigned x) {
    x ^= x >> 16; x *= 0x7feb352du;
    x ^= x >> 15; x *= 0x846ca68bu;
    x ^= x >> 16;
    return x;
}

// ---------------------------------------------------------------------------
// Setup kernel: one block. Builds the const image in d_ws and zeros invarl.
// ---------------------------------------------------------------------------
__global__ __launch_bounds__(256)
void ginvar_setup_kernel(
    const float* __restrict__ adj,
    const float* __restrict__ conv_W, const float* __restrict__ conv_b,
    const float* __restrict__ gate_W, const float* __restrict__ gate_b,
    const float* __restrict__ v1_W, const float* __restrict__ v1_b,
    const float* __restrict__ v2_W, const float* __restrict__ v2_b,
    const float* __restrict__ v3_W, const float* __restrict__ v3_b,
    const float* __restrict__ mu_W, const float* __restrict__ mu_b,
    const float* __restrict__ ls_W, const float* __restrict__ ls_b,
    float* __restrict__ wsf, float* __restrict__ invarl_slot)
{
    const int tid = threadIdx.x;
    _Float16* wsh = (_Float16*)((char*)wsf + F16_BYTE_OFF);

    if (tid == 0) {
        *invarl_slot = 0.0f;
        float a[NN][NN];
        for (int i = 0; i < NN; ++i)
            for (int j = 0; j < NN; ++j)
                a[i][j] = adj[i * NN + j];

        int ones = 0;
        for (int i = 0; i < NN; ++i)
            for (int j = 0; j < NN; ++j)
                if (a[i][j] == 1.0f) ones++;

        // graph cut: drop kcut = max(1, cm/2) triu edges chosen by hash rank
        int cm = 0;
        for (int i = 0; i < NN; ++i)
            for (int j = i; j < NN; ++j)
                if (a[i][j] == 1.0f) cm++;
        int kcut = cm / 2; if (kcut < 1) kcut = 1;

        float acut[NN][NN];
        for (int i = 0; i < NN; ++i)
            for (int j = 0; j < NN; ++j) acut[i][j] = a[i][j];
        for (int i = 0; i < NN; ++i) {
            for (int j = i; j < NN; ++j) {
                if (a[i][j] != 1.0f) continue;
                unsigned mine = hash32(0x9e3779b9u + (unsigned)(i * NN + j));
                int rank = 0;
                for (int p = 0; p < NN; ++p)
                    for (int q = p; q < NN; ++q) {
                        if (a[p][q] != 1.0f || (p == i && q == j)) continue;
                        unsigned o = hash32(0x9e3779b9u + (unsigned)(p * NN + q));
                        if (o < mine || (o == mine && (p * NN + q) < (i * NN + j))) rank++;
                    }
                if (rank < kcut) { acut[i][j] = 0.0f; acut[j][i] = 0.0f; }
            }
        }

        // edge add: add kadd = min(15 - ones, zm) strict-upper missing edges
        int zm = 0;
        for (int i = 0; i < NN; ++i)
            for (int j = i + 1; j < NN; ++j)
                if (a[i][j] == 0.0f) zm++;
        int kadd = 15 - ones; if (kadd > zm) kadd = zm; if (kadd < 0) kadd = 0;

        float aadd[NN][NN];
        for (int i = 0; i < NN; ++i)
            for (int j = 0; j < NN; ++j) aadd[i][j] = a[i][j];
        for (int i = 0; i < NN; ++i) {
            for (int j = i + 1; j < NN; ++j) {
                if (a[i][j] != 0.0f) continue;
                unsigned mine = hash32(0x85ebca6bu + (unsigned)(i * NN + j));
                int rank = 0;
                for (int p = 0; p < NN; ++p)
                    for (int q = p + 1; q < NN; ++q) {
                        if (a[p][q] != 0.0f || (p == i && q == j)) continue;
                        unsigned o = hash32(0x85ebca6bu + (unsigned)(p * NN + q));
                        if (o < mine || (o == mine && (p * NN + q) < (i * NN + j))) rank++;
                    }
                if (rank < kadd) { aadd[i][j] = 1.0f; aadd[j][i] = 1.0f; }
            }
        }

        // sym-normalize: An = D^-1/2 (2*(a!=0) + I) D^-1/2
        for (int v = 0; v < 3; ++v) {
            float (*src)[NN] = (v == 0) ? a : (v == 1) ? acut : aadd;
            float* dst = wsf + ((v == 0) ? OFF_AN : (v == 1) ? OFF_AC : OFF_AA);
            float A2[NN][NN], dis[NN];
            for (int i = 0; i < NN; ++i) {
                float s = 0.0f;
                for (int j = 0; j < NN; ++j) {
                    float val = (src[i][j] != 0.0f) ? 2.0f : 0.0f;
                    if (i == j) val += 1.0f;
                    A2[i][j] = val; s += val;
                }
                dis[i] = rsqrtf(s);
            }
            for (int i = 0; i < NN; ++i)
                for (int j = 0; j < NN; ++j)
                    dst[i * NN + j] = dis[i] * A2[i][j] * dis[j];
        }
    }

    if (tid == 1) {
        for (int i = 0; i < 4; ++i) {
            wsf[OFF_CB + i]  = conv_b[i]; wsf[OFF_V1B + i] = v1_b[i];
            wsf[OFF_V2B + i] = v2_b[i];   wsf[OFF_V3B + i] = v3_b[i];
        }
        for (int i = 0; i < 96; ++i) wsf[OFF_GW + i] = gate_W[i];
        for (int i = 0; i < 3; ++i)  wsf[OFF_GB + i] = gate_b[i];
        for (int i = 0; i < 8; ++i) { wsf[OFF_MUW + i] = mu_W[i]; wsf[OFF_LSW + i] = ls_W[i]; }
        for (int i = 0; i < 2; ++i) { wsf[OFF_MUB + i] = mu_b[i]; wsf[OFF_LSB + i] = ls_b[i]; }
    }

    // pack transposed f16 weight B-operands: WcatT[col][k], WnoiseT[col][k]
    {
        const int col = tid >> 4;        // 0..15
        const int mtx = col >> 2;
        const int hc  = col & 3;
        const float* wsrc = (mtx == 0) ? conv_W : (mtx == 1) ? v1_W
                          : (mtx == 2) ? v2_W : v3_W;
        const int k0 = (tid & 15) * 16;
        for (int k = k0; k < k0 + 16; ++k) {
            float w = wsrc[k * HH + hc];
            wsh[col * DIN + k]               = (_Float16)w;
            wsh[4096 + col * DIN + k]        =
                (_Float16)((mtx == 1) ? NOISE_W_SCALE * w : 0.0f);
        }
    }
}

// ---------------------------------------------------------------------------
// Main fused kernel.
// ---------------------------------------------------------------------------
__global__ __launch_bounds__(256)
void ginvar_enc_fused_kernel(
    const float* __restrict__ nodef,      // [B, 8, 256]
    const float* __restrict__ init_dist,  // [B, 8, 2]
    const float* __restrict__ wsf,        // const image (floats)
    float* __restrict__ out,              // z | mu | logvar | invarl
    int B)
{
    __shared__ __align__(16) _Float16 Wpack[WPACK_HALVES];      // WcatT | WnoiseT
    __shared__ __align__(16) float projls[ROWS_PER_BLOCK * 16]; // WMMA results
    __shared__ float xls[BATCH_PER_BLOCK * 32];
    __shared__ float invls[BATCH_PER_BLOCK * 32];
    __shared__ float vwls[BATCH_PER_BLOCK * 3];
    __shared__ __align__(16) float constls[CONST_FLOATS];
    __shared__ float bacc;

    const int tid = threadIdx.x;
    const size_t Z = (size_t)B * NN * DOUT;

    // ---- Phase 0: vectorized const fill ------------------------------------
    if (tid == 0) bacc = 0.0f;
    for (int i = tid; i < CONST_FLOATS; i += 256) constls[i] = wsf[i];
    {
        const uint4* src = (const uint4*)((const char*)wsf + F16_BYTE_OFF);
        uint4* dst = (uint4*)Wpack;
        #pragma unroll
        for (int i = 0; i < 4; ++i)          // 1024 uint4 = 16 KB
            dst[tid + i * 256] = src[tid + i * 256];
    }
    __syncthreads();

    // ---- Phase 1: fused 4-way projection via WMMA ---------------------------
    // Each wave: D[16,16] = A[16,256] @ [conv|v1|v2|v3]
    //                     + Noise[16,256] @ [0 | (0.1/32)*v1 | 0 | 0]
    {
        const int wave  = tid >> 5;
        const int lane  = tid & 31;
        const int aRow  = blockIdx.x * ROWS_PER_BLOCK + wave * 16 + (lane & 15);
        const int kHalf = (lane >> 4) * 8;      // 16-bit A/B lane layout
        const float* arow = nodef + (size_t)aRow * DIN;
        const _Float16* wc = &Wpack[(lane & 15) * DIN];
        const _Float16* wn = wc + 4096;

        v8f acc = {};
        #pragma unroll
        for (int kc = 0; kc < 8; ++kc) {
            const int k0 = kc * 32 + kHalf;
            __builtin_prefetch((const void*)(arow + k0 + 128), 0, 0);
            // A tile: 16 f32 -> f16 (K = k0..k0+7, k0+16..k0+23)
            float4 f0 = *(const float4*)(arow + k0);
            float4 f1 = *(const float4*)(arow + k0 + 4);
            float4 f2 = *(const float4*)(arow + k0 + 16);
            float4 f3 = *(const float4*)(arow + k0 + 20);
            v16h av;
            av[0]=(_Float16)f0.x; av[1]=(_Float16)f0.y; av[2]=(_Float16)f0.z; av[3]=(_Float16)f0.w;
            av[4]=(_Float16)f1.x; av[5]=(_Float16)f1.y; av[6]=(_Float16)f1.z; av[7]=(_Float16)f1.w;
            av[8]=(_Float16)f2.x; av[9]=(_Float16)f2.y; av[10]=(_Float16)f2.z; av[11]=(_Float16)f2.w;
            av[12]=(_Float16)f3.x; av[13]=(_Float16)f3.y; av[14]=(_Float16)f3.z; av[15]=(_Float16)f3.w;
            // B1 tile from LDS (two ds_load_b128)
            v8h blo = *(const v8h*)(wc + k0);
            v8h bhi = *(const v8h*)(wc + k0 + 16);
            v16h bv;
            #pragma unroll
            for (int j = 0; j < 8; ++j) { bv[j] = blo[j]; bv[8 + j] = bhi[j]; }
            acc = __builtin_amdgcn_wmma_f32_16x16x32_f16(
                false, av, false, bv, (short)0, acc, false, false);
            // Noise A tile: one hash -> two int16 noise values
            v16h nv;
            const unsigned pbase = ((unsigned)aRow * DIN + (unsigned)k0) >> 1;
            #pragma unroll
            for (int p = 0; p < 4; ++p) {
                const unsigned u = hash32(pbase + (unsigned)p);
                nv[2*p]   = (_Float16)((float)(short)(u & 0xffffu) * NOISE_ELEM_SCALE);
                nv[2*p+1] = (_Float16)((float)(short)(u >> 16)     * NOISE_ELEM_SCALE);
            }
            #pragma unroll
            for (int p = 0; p < 4; ++p) {
                const unsigned u = hash32(pbase + 8u + (unsigned)p);
                nv[8+2*p]   = (_Float16)((float)(short)(u & 0xffffu) * NOISE_ELEM_SCALE);
                nv[8+2*p+1] = (_Float16)((float)(short)(u >> 16)     * NOISE_ELEM_SCALE);
            }
            v8h nlo = *(const v8h*)(wn + k0);
            v8h nhi = *(const v8h*)(wn + k0 + 16);
            v16h bn;
            #pragma unroll
            for (int j = 0; j < 8; ++j) { bn[j] = nlo[j]; bn[8 + j] = nhi[j]; }
            acc = __builtin_amdgcn_wmma_f32_16x16x32_f16(
                false, nv, false, bn, (short)0, acc, false, false);
        }
        // D: vgpr r, lanes 0-15 -> M=r, lanes 16-31 -> M=r+8; col = lane&15
        const int mh  = (lane >> 4) * 8;
        const int col = lane & 15;
        #pragma unroll
        for (int r = 0; r < 8; ++r)
            projls[(wave * 16 + r + mh) * 16 + col] = acc[r];
    }
    __syncthreads();

    // ---- Phase 2: graph mixes, gates, heads --------------------------------
    const float* Anls = constls + OFF_AN;
    const float* Acls = constls + OFF_AC;
    const float* Aals = constls + OFF_AA;
    const float* gWls = constls + OFF_GW;

    const int bl  = tid >> 4;
    const int sub = tid & 15;
    const int gb  = blockIdx.x * BATCH_PER_BLOCK + bl;

    float s1e[2], s2e[2], s3e[2];
    #pragma unroll
    for (int t = 0; t < 2; ++t) {
        const int e = sub + t * 16;       // flat (n*4 + h)
        const int n = e >> 2, hc = e & 3;
        float xs = constls[OFF_CB + hc], s1 = constls[OFF_V1B + hc];
        float s2 = constls[OFF_V2B + hc], s3 = constls[OFF_V3B + hc];
        #pragma unroll
        for (int j = 0; j < NN; ++j) {
            const float* pr = &projls[(bl * NN + j) * 16];
            const float an = Anls[n * NN + j];
            xs += an * pr[hc];
            s1 += an * pr[4 + hc];
            s2 += Acls[n * NN + j] * pr[8 + hc];
            s3 += Aals[n * NN + j] * pr[12 + hc];
        }
        xls[bl * 32 + e] = xs;
        s1e[t] = s1; s2e[t] = s2; s3e[t] = s3;
    }
    __syncthreads();

    if (sub == 0) {   // vary-gate softmax over conv features
        float l0 = constls[OFF_GB], l1 = constls[OFF_GB + 1], l2 = constls[OFF_GB + 2];
        for (int e = 0; e < 32; ++e) {
            const float xv = xls[bl * 32 + e];
            l0 += xv * gWls[e * 3]; l1 += xv * gWls[e * 3 + 1]; l2 += xv * gWls[e * 3 + 2];
        }
        const float m = fmaxf(l0, fmaxf(l1, l2));
        const float e0 = expf(l0 - m), e1 = expf(l1 - m), e2 = expf(l2 - m);
        const float inv = 1.0f / (e0 + e1 + e2);
        vwls[bl * 3] = e0 * inv; vwls[bl * 3 + 1] = e1 * inv; vwls[bl * 3 + 2] = e2 * inv;
    }
    __syncthreads();

    const float vw0 = vwls[bl * 3], vw1 = vwls[bl * 3 + 1], vw2 = vwls[bl * 3 + 2];
    #pragma unroll
    for (int t = 0; t < 2; ++t) {
        const int e = sub + t * 16;
        invls[bl * 32 + e] = vw0 * s1e[t] + vw1 * s2e[t] + vw2 * s3e[t];
    }
    __syncthreads();

    {   // mu / logvar / z : sub -> (node, dout)
        const int n = sub >> 1, d = sub & 1;
        float mu = constls[OFF_MUB + d], ls = constls[OFF_LSB + d];
        #pragma unroll
        for (int j = 0; j < NN; ++j) {
            float tm = 0.0f, tl = 0.0f;
            #pragma unroll
            for (int hc = 0; hc < HH; ++hc) {
                const float iv = invls[bl * 32 + j * HH + hc];
                tm += iv * constls[OFF_MUW + hc * DOUT + d];
                tl += iv * constls[OFF_LSW + hc * DOUT + d];
            }
            const float an = Anls[n * NN + j];
            mu += an * tm; ls += an * tl;
        }
        const size_t oi = (size_t)gb * (NN * DOUT) + (size_t)(n * DOUT + d);
        const float z = mu + init_dist[oi] * expf(ls);
        out[oi] = z;
        out[Z + oi] = mu;
        out[2 * Z + oi] = ls;
    }

    if (sub == 0) {   // invariance loss contribution
        float l0 = constls[OFF_GB], l1 = constls[OFF_GB + 1], l2 = constls[OFF_GB + 2];
        for (int e = 0; e < 32; ++e) {
            const float xv = invls[bl * 32 + e];
            l0 += xv * gWls[e * 3]; l1 += xv * gWls[e * 3 + 1]; l2 += xv * gWls[e * 3 + 2];
        }
        const float m = fmaxf(l0, fmaxf(l1, l2));
        const float e0 = expf(l0 - m), e1 = expf(l1 - m), e2 = expf(l2 - m);
        const float inv = 1.0f / (e0 + e1 + e2);
        const float part = fabsf(e0 * inv - vw0) + fabsf(e1 * inv - vw1)
                         + fabsf(e2 * inv - vw2);
        atomicAdd(&bacc, part);   // ds_add_f32
    }
    __syncthreads();
    if (tid == 0)
        atomicAdd(out + 3 * Z, bacc * (1.0f / ((float)B * 3.0f)));
}

extern "C" void kernel_launch(void* const* d_in, const int* in_sizes, int n_in,
                              void* d_out, int out_size, void* d_ws, size_t ws_size,
                              hipStream_t stream) {
    const float* nodef     = (const float*)d_in[0];
    const float* adj       = (const float*)d_in[1];
    const float* init_dist = (const float*)d_in[2];
    const float* conv_W = (const float*)d_in[3];
    const float* conv_b = (const float*)d_in[4];
    const float* gate_W = (const float*)d_in[5];
    const float* gate_b = (const float*)d_in[6];
    const float* v1_W = (const float*)d_in[7];
    const float* v1_b = (const float*)d_in[8];
    const float* v2_W = (const float*)d_in[9];
    const float* v2_b = (const float*)d_in[10];
    const float* v3_W = (const float*)d_in[11];
    const float* v3_b = (const float*)d_in[12];
    const float* mu_W = (const float*)d_in[13];
    const float* mu_b = (const float*)d_in[14];
    const float* ls_W = (const float*)d_in[15];
    const float* ls_b = (const float*)d_in[16];
    float* out = (float*)d_out;
    float* wsf = (float*)d_ws;

    const int B = in_sizes[0] / (NN * DIN);           // 32768
    const size_t Z = (size_t)B * NN * DOUT;

    ginvar_setup_kernel<<<1, 256, 0, stream>>>(
        adj, conv_W, conv_b, gate_W, gate_b,
        v1_W, v1_b, v2_W, v2_b, v3_W, v3_b,
        mu_W, mu_b, ls_W, ls_b,
        wsf, out + 3 * Z);

    const int blocks = (B + BATCH_PER_BLOCK - 1) / BATCH_PER_BLOCK;
    ginvar_enc_fused_kernel<<<blocks, 256, 0, stream>>>(
        nodef, init_dist, wsf, out, B);
}
